// SetConvLayer_45767171506775
// MI455X (gfx1250) — matrix operand order repaired and checked by reading, not compiled, not run on hardware
//
#include <hip/hip_runtime.h>

typedef __attribute__((ext_vector_type(16))) _Float16 v16h;
typedef __attribute__((ext_vector_type(8)))  float    v8f;

#define NPTS 16384
#define NCTR 4096
#define CIN  131
#define R2   0.04f

#define XS_STRIDE 168   // 160 (padded K of layer1) + 8 pad, 16B-aligned rows
#define H_STRIDE  136   // 128 + 8 pad, 16B-aligned rows

// ---------- fragment builders (ISA 7.12.2 layouts, wave32) ----------
// A 16x32 f16: lane half h holds K {k0..k0+7, k0+16..k0+23}, k0 = 8h (+32*kblk)
__device__ __forceinline__ v16h frag_a(const _Float16* p) {
    union { v16h v; uint4 q[2]; } f;
    f.q[0] = *(const uint4*)(p);
    f.q[1] = *(const uint4*)(p + 16);
    return f.v;
}
// B 32x16 f16 (column-major per lane): lane half h holds K {k0..k0+7, k0+16..k0+23}, k0 = 8h
__device__ __forceinline__ v16h frag_b(const _Float16* p) {
    union { v16h v; uint4 q[2]; } f;
    f.q[0] = *(const uint4*)(p);
    f.q[1] = *(const uint4*)(p + 16);
    return f.v;
}

// ---------- prep: f16 transposed weights + compacted float4 positions ----------
__global__ void prep_kernel(const float* __restrict__ x,
                            const float* __restrict__ W1, const float* __restrict__ W2,
                            const float* __restrict__ W3,
                            _Float16* __restrict__ W1t, _Float16* __restrict__ W2t,
                            _Float16* __restrict__ W3t, float4* __restrict__ pos) {
    int i = blockIdx.x * blockDim.x + threadIdx.x;
    if (i < 128 * 160) {                 // W1t: [128][160], K padded 131->160
        int n = i / 160, k = i % 160;
        W1t[i] = (_Float16)(k < CIN ? W1[k * 128 + n] : 0.f);
    }
    if (i < 128 * 128) {                 // W2t: [128][128]
        int n = i / 128, k = i % 128;
        W2t[i] = (_Float16)W2[k * 128 + n];
    }
    if (i < 256 * 128) {                 // W3t: [256][128]
        int n = i / 128, k = i % 128;
        W3t[i] = (_Float16)W3[k * 256 + n];
    }
    if (i < NPTS) {                      // dense positions for FPS/KNN sweeps
        pos[i] = make_float4(x[i * CIN + 0], x[i * CIN + 1], x[i * CIN + 2], 0.f);
    }
}

// ---------- FPS: one block, 1024 threads, 16 pts/thread ----------
__global__ __launch_bounds__(1024) void fps_kernel(const float4* __restrict__ pos,
                                                   int* __restrict__ sel,
                                                   float* __restrict__ centers) {
    int t = threadIdx.x;
    float px[16], py[16], pz[16], md[16];
#pragma unroll
    for (int i = 0; i < 16; ++i) {
        float4 p = pos[i * 1024 + t];
        px[i] = p.x; py[i] = p.y; pz[i] = p.z;
        md[i] = 3.0e38f;
    }
    __shared__ float lx, ly, lz;
    __shared__ float rv[32];
    __shared__ int   ri[32];
    if (t == 0) {
        lx = px[0]; ly = py[0]; lz = pz[0];       // point 0 is first selection
        sel[0] = 0;
        centers[0] = px[0]; centers[1] = py[0]; centers[2] = pz[0];
    }
    __syncthreads();
    for (int k = 1; k < NCTR; ++k) {
        float cx = lx, cy = ly, cz = lz;
        float bv = -1.f; int bi = 0x7fffffff;
#pragma unroll
        for (int i = 0; i < 16; ++i) {
            float dx = px[i] - cx, dy = py[i] - cy, dz = pz[i] - cz;
            float d  = dx * dx + dy * dy + dz * dz;
            float nm = fminf(md[i], d);
            md[i] = nm;
            int p = i * 1024 + t;
            if (nm > bv || (nm == bv && p < bi)) { bv = nm; bi = p; }
        }
#pragma unroll
        for (int off = 16; off >= 1; off >>= 1) {   // wave32 reduce
            float ov = __shfl_xor(bv, off, 32);
            int   oi = __shfl_xor(bi, off, 32);
            if (ov > bv || (ov == bv && oi < bi)) { bv = ov; bi = oi; }
        }
        if ((t & 31) == 0) { rv[t >> 5] = bv; ri[t >> 5] = bi; }
        __syncthreads();
        if (t < 32) {
            float v2 = rv[t]; int i2 = ri[t];
#pragma unroll
            for (int off = 16; off >= 1; off >>= 1) {
                float ov = __shfl_xor(v2, off, 32);
                int   oi = __shfl_xor(i2, off, 32);
                if (ov > v2 || (ov == v2 && oi < i2)) { v2 = ov; i2 = oi; }
            }
            if (t == 0) {
                sel[k] = i2;
                float4 a = pos[i2];
                lx = a.x; ly = a.y; lz = a.z;
                centers[k * 3 + 0] = a.x; centers[k * 3 + 1] = a.y; centers[k * 3 + 2] = a.z;
            }
        }
        __syncthreads();
    }
}

// ---------- neighbor select: radius compact + rank top-32 ----------
__global__ __launch_bounds__(256) void knn_kernel(const float4* __restrict__ pos,
                                                  const float* __restrict__ centers,
                                                  int* __restrict__ nbr,
                                                  int* __restrict__ cnt) {
    int c = blockIdx.x, t = threadIdx.x;
    __shared__ int   sc;
    __shared__ int   sidx[256];
    __shared__ float sd2[256];
    if (t == 0) sc = 0;
    __syncthreads();
    float cx = centers[c * 3], cy = centers[c * 3 + 1], cz = centers[c * 3 + 2];
    for (int i = t; i < NPTS; i += 256) {
        float4 p = pos[i];
        float dx = p.x - cx, dy = p.y - cy, dz = p.z - cz;
        float d2 = dx * dx + dy * dy + dz * dz;
        if (d2 <= R2) {
            int q = atomicAdd(&sc, 1);
            if (q < 256) { sidx[q] = i; sd2[q] = d2; }
        }
    }
    __syncthreads();
    int n = sc; if (n > 256) n = 256;
    if (n <= 32) {
        if (t < n) nbr[c * 32 + t] = sidx[t];
        if (t == 0) cnt[c] = n;
    } else {
        if (t < n) {
            float dm = sd2[t]; int im = sidx[t];
            int rank = 0;
            for (int j = 0; j < n; ++j) {
                float dj = sd2[j]; int ij = sidx[j];
                if (dj < dm || (dj == dm && ij < im)) rank++;
            }
            if (rank < 32) nbr[c * 32 + rank] = im;
        }
        if (t == 0) cnt[c] = 32;
    }
}

// ---------- one GEMM layer on LDS tiles (wave owns 32 rows = 2 M-tiles) ----------
__device__ __forceinline__ void gemm_layer(const _Float16* in_lds, int in_stride,
                                           const _Float16* __restrict__ wt, int wt_stride,
                                           const float* __restrict__ bias,
                                           int nblocks, int kblocks,
                                           _Float16* out_lds, int out_stride,
                                           int w, int lane) {
    int half = lane >> 4, lrow = lane & 15;
    int rbase = w * 32;
    for (int n = 0; n < nblocks; ++n) {
        float bb = bias[n * 16 + lrow];
        v8f acc0 = {}, acc1 = {};
        const _Float16* wr = wt + (n * 16 + lrow) * wt_stride;
        for (int k = 0; k < kblocks; ++k) {
            v16h bf = frag_b(wr + k * 32 + half * 8);
            v16h a0 = frag_a(in_lds + (rbase + lrow) * in_stride + k * 32 + half * 8);
            v16h a1 = frag_a(in_lds + (rbase + 16 + lrow) * in_stride + k * 32 + half * 8);
            acc0 = __builtin_amdgcn_wmma_f32_16x16x32_f16(false, a0, false, bf, (short)0, acc0, false, false);
            acc1 = __builtin_amdgcn_wmma_f32_16x16x32_f16(false, a1, false, bf, (short)0, acc1, false, false);
        }
#pragma unroll
        for (int r = 0; r < 8; ++r) {
            int row0 = rbase + r + half * 8;     // D: row = r + 8*half, col = lane%16
            out_lds[row0 * out_stride + n * 16 + lrow]        = (_Float16)fmaxf(acc0[r] + bb, 0.f);
            out_lds[(row0 + 16) * out_stride + n * 16 + lrow] = (_Float16)fmaxf(acc1[r] + bb, 0.f);
        }
    }
}

// ---------- gather + 3-layer WMMA MLP + masked max-pool ----------
__global__ __launch_bounds__(256) void mlp_kernel(const float* __restrict__ x,
                                                  const float* __restrict__ centers,
                                                  const int* __restrict__ nbr,
                                                  const int* __restrict__ cntb,
                                                  const _Float16* __restrict__ W1t,
                                                  const _Float16* __restrict__ W2t,
                                                  const _Float16* __restrict__ W3t,
                                                  const float* __restrict__ b1,
                                                  const float* __restrict__ b2,
                                                  const float* __restrict__ b3,
                                                  float* __restrict__ out) {
    __shared__ __align__(16) _Float16 smemA[256 * XS_STRIDE];  // X, reused as H2
    __shared__ __align__(16) _Float16 smemB[256 * H_STRIDE];   // H1
    _Float16* Xs = smemA;
    _Float16* H1 = smemB;
    _Float16* H2 = smemA;

    int t = threadIdx.x;
    int w = t >> 5, lane = t & 31, half = lane >> 4, lrow = lane & 15;
    int c = blockIdx.x * 8 + w;                  // wave w owns center c (32 rows)
    int cnt = cntb[c];
    float ccx = centers[c * 3], ccy = centers[c * 3 + 1], ccz = centers[c * 3 + 2];

    // gather: rows = slots; cols 0..127 feat, 128..130 rel pos, 131..159 zero
    // packed: each lane writes an f16x2 dword to LDS
    for (int s = 0; s < 32; ++s) {
        int j = (s < cnt) ? nbr[c * 32 + s] : 0;
        const float* xr = x + j * CIN;
        int row = w * 32 + s;
#pragma unroll
        for (int st = 0; st < 3; ++st) {
            int col = (st * 32 + lane) * 2;
            if (col < 160) {
                float v0, v1;
                if (col < 128)       { v0 = xr[3 + col]; v1 = xr[4 + col]; }
                else if (col == 128) { v0 = xr[0] - ccx; v1 = xr[1] - ccy; }
                else if (col == 130) { v0 = xr[2] - ccz; v1 = 0.f; }
                else                 { v0 = 0.f;         v1 = 0.f; }
                union { _Float16 h[2]; unsigned u; } pk;
                pk.h[0] = (_Float16)v0; pk.h[1] = (_Float16)v1;
                *(unsigned*)&Xs[row * XS_STRIDE + col] = pk.u;
            }
        }
    }
    __syncthreads();

    gemm_layer(Xs, XS_STRIDE, W1t, 160, b1, 8, 5, H1, H_STRIDE, w, lane);   // 160->128
    __syncthreads();
    gemm_layer(H1, H_STRIDE, W2t, 128, b2, 8, 4, H2, H_STRIDE, w, lane);    // 128->128
    __syncthreads();

    // layer 3 (128->256) fused with masked max-pool over 32 slots
    int rbase = w * 32;
    for (int n = 0; n < 16; ++n) {
        float bb = b3[n * 16 + lrow];
        v8f acc0 = {}, acc1 = {};
        const _Float16* wr = W3t + (n * 16 + lrow) * 128;
        for (int k = 0; k < 4; ++k) {
            v16h bf = frag_b(wr + k * 32 + half * 8);
            v16h a0 = frag_a(H2 + (rbase + lrow) * H_STRIDE + k * 32 + half * 8);
            v16h a1 = frag_a(H2 + (rbase + 16 + lrow) * H_STRIDE + k * 32 + half * 8);
            acc0 = __builtin_amdgcn_wmma_f32_16x16x32_f16(false, a0, false, bf, (short)0, acc0, false, false);
            acc1 = __builtin_amdgcn_wmma_f32_16x16x32_f16(false, a1, false, bf, (short)0, acc1, false, false);
        }
        float m = -3.0e38f;
#pragma unroll
        for (int r = 0; r < 8; ++r) {
            int s0 = r + half * 8;               // slot of acc0[r]; acc1 -> s0+16
            float v0 = fmaxf(acc0[r] + bb, 0.f);
            float v1 = fmaxf(acc1[r] + bb, 0.f);
            if (s0 < cnt)      m = fmaxf(m, v0);
            if (s0 + 16 < cnt) m = fmaxf(m, v1);
        }
        m = fmaxf(m, __shfl_xor(m, 16, 32));     // merge lane halves (same col)
        if (half == 0) out[c * 256 + n * 16 + lrow] = (cnt > 0) ? m : 0.f;
    }
}

extern "C" void kernel_launch(void* const* d_in, const int* in_sizes, int n_in,
                              void* d_out, int out_size, void* d_ws, size_t ws_size,
                              hipStream_t stream) {
    const float* x  = (const float*)d_in[0];
    const float* W1 = (const float*)d_in[1];
    const float* b1 = (const float*)d_in[2];
    const float* W2 = (const float*)d_in[3];
    const float* b2 = (const float*)d_in[4];
    const float* W3 = (const float*)d_in[5];
    const float* b3 = (const float*)d_in[6];
    float* out = (float*)d_out;

    char* ws = (char*)d_ws;
    _Float16* W1t = (_Float16*)(ws + 0);        // 128*160*2 = 40960
    _Float16* W2t = (_Float16*)(ws + 40960);    // 128*128*2 = 32768
    _Float16* W3t = (_Float16*)(ws + 73728);    // 256*128*2 = 65536
    int*      sel = (int*)(ws + 139264);        // 4096*4
    float*    cen = (float*)(ws + 155648);      // 4096*3*4
    int*      nbr = (int*)(ws + 204800);        // 4096*32*4
    int*      cnt = (int*)(ws + 729088);        // 4096*4
    float4*   pos = (float4*)(ws + 745472);     // 16384*16 (total 1,007,616 B)

    prep_kernel<<<128, 256, 0, stream>>>(x, W1, W2, W3, W1t, W2t, W3t, pos);
    fps_kernel<<<1, 1024, 0, stream>>>(pos, sel, cen);
    knn_kernel<<<NCTR, 256, 0, stream>>>(pos, cen, nbr, cnt);
    mlp_kernel<<<NCTR / 8, 256, 0, stream>>>(x, cen, nbr, cnt, W1t, W2t, W3t, b1, b2, b3, out);
}
